// SubstitutionHead_88029649699354
// MI455X (gfx1250) — matrix operand order repaired and checked by reading, not compiled, not run on hardware
//
#include <hip/hip_runtime.h>
#include <hip/hip_bf16.h>
#include <stdint.h>

// ---------------------------------------------------------------------------
// SubstitutionHead on MI455X (gfx1250, wave32).
//   convert: x fp32 -> (A_hi, A_lo) bf16 split-precision pair
//   GEMM1:   [8192,512] @ [512,1024] via bf16x3 WMMA (K=32/issue), fp32 acc
//   scan:    per-batch argmax(depth) + ballot prefix -> src, cnt
//   GEMM2:   gather(y1)[32768,256] @ Wc[256,64] (+bc) via fp32 WMMA -> d_out
// Wc = fold(W0, Wl), bc = b0@Wl^T + bl  (deconv0 + linear fused, 8x fewer FLOPs)
// ---------------------------------------------------------------------------

typedef float  v2f   __attribute__((ext_vector_type(2)));
typedef float  v8f   __attribute__((ext_vector_type(8)));
typedef __bf16 bf16_t;
typedef __bf16 v8bf  __attribute__((ext_vector_type(8)));
typedef __bf16 v16bf __attribute__((ext_vector_type(16)));

#define NB     4
#define E2     256
#define E4     128
#define CS     4
#define S_DIM  24576
#define V_DIM  16
#define L1     8192      // T*CS
#define M1     8192      // N*T
#define K1     512
#define N1     1024      // CS*E2
#define M2     32768     // NB*L1
#define K2     256
#define N2     64        // CS*V

#define APAD   40        // LDS row length (bf16) for 32-K tiles: 80B = 20 banks
                         // 20*m mod 64 distinct for m=0..15 -> conflict-free b128

static __device__ __forceinline__ v16bf cat8(v8bf a, v8bf b) {
    return __builtin_shufflevector(a, b, 0,1,2,3,4,5,6,7,8,9,10,11,12,13,14,15);
}

// ---------------------------------------------------------------------------
// x fp32 -> hi/lo bf16 split pair (hi = RNE(x), lo = RNE(x - hi))
__global__ __launch_bounds__(256)
void convert_x(const float* __restrict__ x, bf16_t* __restrict__ Ah,
               bf16_t* __restrict__ Al) {
    int idx = blockIdx.x * 256 + threadIdx.x;       // M1*K1
    if (idx < M1 * K1) {
        float v = x[idx];
        bf16_t h = (bf16_t)v;
        Ah[idx] = h;
        Al[idx] = (bf16_t)(v - (float)h);
    }
}

// B1 transposed+split: Bt[n, i] = W1[i, o, k], n = k*256+o  (layout [N1][K1])
__global__ __launch_bounds__(256)
void permute_w1(const float* __restrict__ W1, bf16_t* __restrict__ Bh,
                bf16_t* __restrict__ Bl) {
    int idx = blockIdx.x * 256 + threadIdx.x;       // N1*K1
    if (idx < N1 * K1) {
        int n = idx >> 9;           // 0..1023
        int i = idx & 511;
        int k = n >> 8;
        int o = n & 255;
        float v = W1[i * 1024 + o * 4 + k];
        bf16_t h = (bf16_t)v;
        Bh[idx] = h;
        Bl[idx] = (bf16_t)(v - (float)h);
    }
}

// Wc[i, k*16+v] = sum_o W0[i,o,k] * Wl[v,o];  bc[v] = sum_o b0[o]*Wl[v,o] + bl[v]
__global__ __launch_bounds__(256)
void fold_w0(const float* __restrict__ W0, const float* __restrict__ Wl,
             const float* __restrict__ b0, const float* __restrict__ bl,
             float* __restrict__ Wc, float* __restrict__ bc) {
    int idx = blockIdx.x * 256 + threadIdx.x;       // 256*64
    if (idx < K2 * N2) {
        int i = idx >> 6;
        int c = idx & 63;
        int k = c >> 4;
        int v = c & 15;
        float s = 0.f;
        #pragma unroll 4
        for (int o = 0; o < E4; ++o)
            s += W0[i * (E4 * CS) + o * CS + k] * Wl[v * E4 + o];
        Wc[idx] = s;
        if (idx < V_DIM) {
            float t = bl[idx];
            for (int o = 0; o < E4; ++o) t += b0[o] * Wl[idx * E4 + o];
            bc[idx] = t;
        }
    }
}

// ---------------------------------------------------------------------------
// One block per batch row: argmax(depth[n,:]) then ballot prefix-compaction.
__global__ __launch_bounds__(1024)
void scan_compact(const int* __restrict__ value, const int* __restrict__ depth,
                  int* __restrict__ src, int* __restrict__ cnt) {
    const int n    = blockIdx.x;
    const int tid  = threadIdx.x;
    const int lane = tid & 31;
    const int wave = tid >> 5;

    __shared__ int smax[32], sidx[32], swsum[33];
    __shared__ int idx1_s, chunkBase;

    int bestv = INT32_MIN, besti = S_DIM;
    for (int t = tid; t < S_DIM; t += 1024) {
        int d = depth[n * S_DIM + t];
        if (d > bestv || (d == bestv && t < besti)) { bestv = d; besti = t; }
    }
    #pragma unroll
    for (int off = 16; off > 0; off >>= 1) {
        int ov = __shfl_down(bestv, off, 32);
        int oi = __shfl_down(besti, off, 32);
        if (ov > bestv || (ov == bestv && oi < besti)) { bestv = ov; besti = oi; }
    }
    if (lane == 0) { smax[wave] = bestv; sidx[wave] = besti; }
    __syncthreads();
    if (tid == 0) {
        int bv = smax[0], bi = sidx[0];
        for (int w = 1; w < 32; ++w)
            if (smax[w] > bv || (smax[w] == bv && sidx[w] < bi)) { bv = smax[w]; bi = sidx[w]; }
        idx1_s = bi;
        chunkBase = 0;
    }
    __syncthreads();
    const int idx1 = idx1_s;

    for (int c0 = 0; c0 < L1; c0 += 1024) {
        int t = c0 + tid;
        int m = (t < idx1 && value[n * S_DIM + t] == 2) ? 1 : 0;
        unsigned bal = (unsigned)__ballot(m);               // wave32: low 32 bits
        int wprefix = __popc(bal & ((1u << lane) - 1u));
        if (lane == 0) swsum[wave] = __popc(bal);
        __syncthreads();
        if (tid == 0) {
            int s = 0;
            for (int w = 0; w < 32; ++w) { int x = swsum[w]; swsum[w] = s; s += x; }
            swsum[32] = s;
        }
        __syncthreads();
        if (m) src[n * L1 + chunkBase + swsum[wave] + wprefix] = t;
        __syncthreads();
        if (tid == 0) chunkBase += swsum[32];
        __syncthreads();
    }
    if (tid == 0) cnt[n] = chunkBase;
}

// ---------------------------------------------------------------------------
// GEMM1 (bf16x3): Y[8192,1024] = x @ B1 + b1[col%256], fp32 accumulate.
// acc += Ahi*Bhi + Ahi*Blo + Alo*Bhi   (three v_wmma_f32_16x16x32_bf16)
// Block tile 64x256, BK=32. 8 waves as 2x4; wave tile 32x64 = 2x4 wmma tiles.
// B is pre-transposed [n][k] so each lane's two 8-K runs are contiguous in LDS.
__global__ __launch_bounds__(256)
void gemm1_wmma(const bf16_t* __restrict__ Ah, const bf16_t* __restrict__ Al,
                const bf16_t* __restrict__ Bh, const bf16_t* __restrict__ Bl,
                const float* __restrict__ b1, float* __restrict__ Y) {
    __shared__ __attribute__((aligned(16))) bf16_t As[2][64][APAD];
    __shared__ __attribute__((aligned(16))) bf16_t Bs[2][256][APAD];
    const int tid  = threadIdx.x;
    const int wave = tid >> 5;
    const int lane = tid & 31;
    const int wr   = wave >> 2;        // 0..1
    const int wc   = wave & 3;         // 0..3
    const int l16  = lane & 15;
    const int lh   = lane >> 4;        // 0/1
    const int blockM = blockIdx.x * 64;
    const int blockN = blockIdx.y * 256;

    const v8f vzero = {0.f, 0.f, 0.f, 0.f, 0.f, 0.f, 0.f, 0.f};
    v8f acc[2][4];
    #pragma unroll
    for (int i = 0; i < 2; ++i)
        #pragma unroll
        for (int j = 0; j < 4; ++j) acc[i][j] = vzero;

    const int ar = tid >> 2;           // A-tile row this thread loads
    const int ac = (tid & 3) << 3;     // A-tile k-chunk (8 bf16)
    const bf16_t* ah = Ah + (size_t)(blockM + ar) * K1 + ac;
    const bf16_t* al = Al + (size_t)(blockM + ar) * K1 + ac;
    const bf16_t* bh = Bh + (size_t)(blockN + tid) * K1;
    const bf16_t* bl = Bl + (size_t)(blockN + tid) * K1;

    for (int k0 = 0; k0 < K1; k0 += 32) {
        // A tiles (hi/lo): 64x32 bf16, one v8bf per thread per tensor
        *reinterpret_cast<v8bf*>(&As[0][ar][ac]) = *reinterpret_cast<const v8bf*>(ah + k0);
        *reinterpret_cast<v8bf*>(&As[1][ar][ac]) = *reinterpret_cast<const v8bf*>(al + k0);
        // B tiles (hi/lo): 256x32 bf16, one full row (4 v8bf) per thread per tensor
        #pragma unroll
        for (int q = 0; q < 4; ++q) {
            *reinterpret_cast<v8bf*>(&Bs[0][tid][q * 8]) =
                *reinterpret_cast<const v8bf*>(bh + k0 + q * 8);
            *reinterpret_cast<v8bf*>(&Bs[1][tid][q * 8]) =
                *reinterpret_cast<const v8bf*>(bl + k0 + q * 8);
        }
        if (k0 + 32 < K1) {                 // next-tile prefetch (global_prefetch_b8)
            __builtin_prefetch(ah + k0 + 32, 0, 1);
            __builtin_prefetch(bh + k0 + 32, 0, 1);
        }
        __syncthreads();

        // A frags: 16-bit A layout -> per-lane K runs {lh*8..+7, 16+lh*8..+7}
        v16bf afr[2][2];                    // [hi/lo][i]
        #pragma unroll
        for (int h = 0; h < 2; ++h)
            #pragma unroll
            for (int i = 0; i < 2; ++i) {
                int m = wr * 32 + i * 16 + l16;
                v8bf p0 = *reinterpret_cast<const v8bf*>(&As[h][m][lh * 8]);
                v8bf p1 = *reinterpret_cast<const v8bf*>(&As[h][m][16 + lh * 8]);
                afr[h][i] = cat8(p0, p1);
            }
        #pragma unroll
        for (int j = 0; j < 4; ++j) {
            int nn = wc * 64 + j * 16 + l16;
            // B frag: per-lane K runs {lh*8..+7, 16+lh*8..+7} (pre-transposed rows)
            v8bf q0 = *reinterpret_cast<const v8bf*>(&Bs[0][nn][lh * 8]);
            v8bf q1 = *reinterpret_cast<const v8bf*>(&Bs[0][nn][16 + lh * 8]);
            v16bf bfh = cat8(q0, q1);
            v8bf q2 = *reinterpret_cast<const v8bf*>(&Bs[1][nn][lh * 8]);
            v8bf q3 = *reinterpret_cast<const v8bf*>(&Bs[1][nn][16 + lh * 8]);
            v16bf bfl = cat8(q2, q3);
            #pragma unroll
            for (int i = 0; i < 2; ++i) {
                acc[i][j] = __builtin_amdgcn_wmma_f32_16x16x32_bf16(
                    false, afr[0][i], false, bfh, (short)0, acc[i][j], false, false);
                acc[i][j] = __builtin_amdgcn_wmma_f32_16x16x32_bf16(
                    false, afr[0][i], false, bfl, (short)0, acc[i][j], false, false);
                acc[i][j] = __builtin_amdgcn_wmma_f32_16x16x32_bf16(
                    false, afr[1][i], false, bfh, (short)0, acc[i][j], false, false);
            }
        }
        __syncthreads();
    }
    // epilogue: C/D layout — lane%16 = N, VGPR r -> M = r + 8*(lane/16)
    #pragma unroll
    for (int i = 0; i < 2; ++i)
        #pragma unroll
        for (int j = 0; j < 4; ++j) {
            int nn   = blockN + wc * 64 + j * 16 + l16;
            float bv = b1[nn & (E2 - 1)];
            #pragma unroll
            for (int r = 0; r < 8; ++r) {
                int m = blockM + wr * 32 + i * 16 + lh * 8 + r;
                Y[(size_t)m * N1 + nn] = acc[i][j][r] + bv;
            }
        }
}

// ---------------------------------------------------------------------------
// GEMM2 (fp32 WMMA): Out[32768,64] = gather(y1)[32768,256] @ Wc[256,64] + bc.
// Block tile 64x64, BK=16. 8 waves as 4x2; wave tile 16x32 = 1x2 wmma tiles.
__global__ __launch_bounds__(256)
void gemm2_wmma(const float* __restrict__ Y1, const int* __restrict__ src,
                const int* __restrict__ cnt, const float* __restrict__ Wc,
                const float* __restrict__ bc, float* __restrict__ Out) {
    __shared__ float As[64][17];
    __shared__ float Bs[16][68];
    const int tid  = threadIdx.x;
    const int wave = tid >> 5;
    const int lane = tid & 31;
    const int wrow = wave >> 1;        // 0..3
    const int wcol = wave & 1;         // 0..1
    const int l16  = lane & 15;
    const int lh   = lane >> 4;
    const int blockRow = blockIdx.x * 64;        // 8192 % 64 == 0 -> no batch straddle
    const int nb       = blockRow >> 13;
    const int rowInB   = blockRow & (L1 - 1);
    const int count    = cnt[nb];

    const int ar = tid >> 2;                     // 0..63
    const int ac = (tid & 3) << 2;               // 0,4,8,12
    const int jrow = rowInB + ar;
    const bool valid = (jrow < count);
    const int trow = valid ? src[nb * L1 + jrow] : 0;
    const float* arow = Y1 + ((size_t)nb * L1 + trow) * K2;

    const v8f vzero = {0.f, 0.f, 0.f, 0.f, 0.f, 0.f, 0.f, 0.f};
    v8f acc[2] = {vzero, vzero};

    for (int k0 = 0; k0 < K2; k0 += 16) {
        {
            float4 v = make_float4(0.f, 0.f, 0.f, 0.f);
            if (valid) v = *reinterpret_cast<const float4*>(arow + k0 + ac);
            As[ar][ac + 0] = v.x; As[ar][ac + 1] = v.y; As[ar][ac + 2] = v.z; As[ar][ac + 3] = v.w;
        }
        {
            int r = tid >> 4;
            int c = (tid & 15) << 2;
            float4 v = *reinterpret_cast<const float4*>(Wc + (size_t)(k0 + r) * N2 + c);
            Bs[r][c + 0] = v.x; Bs[r][c + 1] = v.y; Bs[r][c + 2] = v.z; Bs[r][c + 3] = v.w;
        }
        __syncthreads();
        #pragma unroll
        for (int kk = 0; kk < 16; kk += 4) {
            const int kb = kk + lh * 2;  // f32 A frag: lanes 0-15 K=kk,kk+1; 16-31 K=kk+2,kk+3
            v2f a;
            a.x = As[wrow * 16 + l16][kb];
            a.y = As[wrow * 16 + l16][kb + 1];
            #pragma unroll
            for (int j = 0; j < 2; ++j) {
                int nn = wcol * 32 + j * 16 + l16;
                v2f b; b.x = Bs[kb][nn]; b.y = Bs[kb + 1][nn];
                acc[j] = __builtin_amdgcn_wmma_f32_16x16x4_f32(
                    false, a, false, b, (short)0, acc[j], false, false);
            }
        }
        __syncthreads();
    }
    #pragma unroll
    for (int j = 0; j < 2; ++j) {
        int nn   = wcol * 32 + j * 16 + l16;
        float bv = bc[nn & (V_DIM - 1)];
        #pragma unroll
        for (int r = 0; r < 8; ++r) {
            int m = blockRow + wrow * 16 + lh * 8 + r;
            Out[(size_t)m * N2 + nn] = acc[j][r] + bv;
        }
    }
}

// ---------------------------------------------------------------------------
extern "C" void kernel_launch(void* const* d_in, const int* in_sizes, int n_in,
                              void* d_out, int out_size, void* d_ws, size_t ws_size,
                              hipStream_t stream) {
    const float* x     = (const float*)d_in[0];   // [4,2048,512]
    const int*   value = (const int*)d_in[1];     // [4,24576]
    const int*   depth = (const int*)d_in[2];     // [4,24576]
    // d_in[3] = pos, unused by the reference output
    const float* W1 = (const float*)d_in[4];      // [512,256,4]
    const float* b1 = (const float*)d_in[5];      // [256]
    const float* W0 = (const float*)d_in[6];      // [256,128,4]
    const float* b0 = (const float*)d_in[7];      // [128]
    const float* Wl = (const float*)d_in[8];      // [16,128]
    const float* bl = (const float*)d_in[9];      // [16]
    float* out = (float*)d_out;                   // [4,32768,16]

    char* ws = (char*)d_ws;
    float*  y1  = (float*) (ws + 0);               // 33,554,432 B  [4,8192,256] fp32
    bf16_t* Ah  = (bf16_t*)(ws + 33554432);        //  8,388,608 B  [8192,512] bf16
    bf16_t* Al  = (bf16_t*)(ws + 41943040);        //  8,388,608 B
    bf16_t* Bh  = (bf16_t*)(ws + 50331648);        //  1,048,576 B  [1024,512] bf16 (B^T)
    bf16_t* Bl  = (bf16_t*)(ws + 51380224);        //  1,048,576 B
    float*  Wc  = (float*) (ws + 52428800);        //     65,536 B  [256,64]
    float*  bc  = (float*) (ws + 52494336);        //        256 B  [16] (padded)
    int*    src = (int*)   (ws + 52494592);        //    131,072 B  [4,8192]
    int*    cnt = (int*)   (ws + 52625664);        //         16 B  [4]

    convert_x   <<<(M1 * K1) / 256, 256, 0, stream>>>(x, Ah, Al);
    permute_w1  <<<(N1 * K1) / 256, 256, 0, stream>>>(W1, Bh, Bl);
    fold_w0     <<<64,  256, 0, stream>>>(W0, Wl, b0, bl, Wc, bc);
    scan_compact<<<NB, 1024, 0, stream>>>(value, depth, src, cnt);
    gemm1_wmma  <<<dim3(M1 / 64, N1 / 256), 256, 0, stream>>>(Ah, Al, Bh, Bl, b1, y1);
    gemm2_wmma  <<<M2 / 64, 256, 0, stream>>>(y1, src, cnt, Wc, bc, out);
}
